// TwoDGAT_12876311954188
// MI455X (gfx1250) — matrix operand order repaired and checked by reading, not compiled, not run on hardware
//
#include <hip/hip_runtime.h>
#include <hip/hip_fp16.h>

#define N_NODES 50000
#define N_EDGES 800000
#define ETOT    (N_EDGES + N_NODES)   // with self loops
#define HID     128                   // 4 heads * 32
#define CH      32                    // per-head channels

typedef __attribute__((ext_vector_type(16))) _Float16 v16h;
typedef __attribute__((ext_vector_type(8)))  _Float16 v8h;
typedef __attribute__((ext_vector_type(8)))  float    v8f;

// ---------- order-preserving float<->uint for atomic max ----------
__device__ __forceinline__ unsigned enc_f32(float f) {
    unsigned u = __float_as_uint(f);
    return (u & 0x80000000u) ? ~u : (u | 0x80000000u);
}
__device__ __forceinline__ float dec_f32(unsigned u) {
    return (u & 0x80000000u) ? __uint_as_float(u & 0x7FFFFFFFu)
                             : __uint_as_float(~u);
}

__device__ __forceinline__ void edge_ends(const int* __restrict__ ei, int e,
                                          int& s, int& d) {
    if (e < N_EDGES) { s = ei[e]; d = ei[N_EDGES + e]; }
    else             { s = d = e - N_EDGES; }           // self loop
}

// ---------- f32 -> f16 pack ----------
__global__ void f32_to_f16(const float* __restrict__ in,
                           _Float16* __restrict__ out, int n) {
    const int idx = blockIdx.x * blockDim.x + threadIdx.x;
    if (idx < n) out[idx] = (_Float16)in[idx];
}

// ---------- WMMA GEMM: Y[N_NODES, NT*16] = Xh[N_NODES, K] @ Wh[K, NT*16] ----------
// One wave computes a 16 x (NT*16) output strip: A fragment shared by NT WMMAs.
// A fragment: packed-f16 16-byte loads; B: strided per-element f16 loads (no per-element converts).
template<int NT>
__global__ void gemm_wmma_f16(const _Float16* __restrict__ Xh,
                              const _Float16* __restrict__ Wh,
                              float* __restrict__ Y, int K) {
    const int lane = threadIdx.x;                       // 0..31
    const int mt   = blockIdx.x * blockDim.y + threadIdx.y;
    if (mt >= N_NODES / 16) return;                     // wave-uniform exit
    const int half = lane >> 4;                         // 0 or 1
    const int row  = mt * 16 + (lane & 15);             // A-matrix M index
    const int Ncols = NT * 16;

    v8f acc[NT] = {};
    for (int kc = 0; kc < K; kc += 32) {
        // A 16x32 f16: lanes 0-15 hold K {0..7,16..23}, lanes 16-31 {8..15,24..31}
        v16h a;
        {
            const v8h lo = *(const v8h*)(Xh + (size_t)row * K + kc + half * 8);
            const v8h hi = *(const v8h*)(Xh + (size_t)row * K + kc + 16 + half * 8);
            #pragma unroll
            for (int i = 0; i < 8; ++i) { a[i] = lo[i]; a[8 + i] = hi[i]; }
        }
        #pragma unroll
        for (int t = 0; t < NT; ++t) {
            // B 32x16 f16: lane&15 = n, elements = K {0..7,16..23} / {8..15,24..31} per half
            v16h b;
            #pragma unroll
            for (int i = 0; i < 8; ++i) {
                b[i]     = Wh[(size_t)(kc + half * 8 + i) * Ncols + t * 16 + (lane & 15)];
                b[8 + i] = Wh[(size_t)(kc + 16 + half * 8 + i) * Ncols + t * 16 + (lane & 15)];
            }
            acc[t] = __builtin_amdgcn_wmma_f32_16x16x32_f16(
                         false, a, false, b, (short)0, acc[t], false, false);
        }
    }
    #pragma unroll
    for (int t = 0; t < NT; ++t)
        #pragma unroll
        for (int r = 0; r < 8; ++r)   // C/D: VGPR r -> M = r + 8*half, N = lane&15
            Y[(size_t)(mt * 16 + half * 8 + r) * Ncols + t * 16 + (lane & 15)] = acc[t][r];
}

// ---------- per-node attention coefficients s,d ----------
__global__ void attn_coef(const float* __restrict__ hlin,
                          const float* __restrict__ a_src,
                          const float* __restrict__ a_dst,
                          float* __restrict__ s, float* __restrict__ d, int H) {
    const int idx = blockIdx.x * blockDim.x + threadIdx.x;
    if (idx >= N_NODES * H) return;
    const int n = idx / H, h = idx % H;
    const float* hp = hlin + (size_t)n * H * CH + h * CH;
    float ss = 0.f, dd = 0.f;
    #pragma unroll
    for (int c = 0; c < CH; ++c) {
        const float v = hp[c];
        ss += v * a_src[h * CH + c];
        dd += v * a_dst[h * CH + c];
    }
    s[idx] = ss; d[idx] = dd;
}

// ---------- init aggregation buffer to bias; init max/den ----------
__global__ void init_agg(float* __restrict__ agg, const float* __restrict__ bias,
                         int OUTD) {
    const int idx = blockIdx.x * blockDim.x + threadIdx.x;
    if (idx >= N_NODES * OUTD) return;
    agg[idx] = bias[idx % OUTD];
}
__global__ void init_mden(unsigned* __restrict__ m, float* __restrict__ den, int H) {
    const int idx = blockIdx.x * blockDim.x + threadIdx.x;
    if (idx >= N_NODES * H) return;
    m[idx] = enc_f32(-INFINITY);
    den[idx] = 0.f;
}

// ---------- edge pass A: leaky-relu logits + segment max ----------
__global__ void edge_lrelu_max(const int* __restrict__ ei,
                               const float* __restrict__ s,
                               const float* __restrict__ d,
                               float* __restrict__ eraw,
                               unsigned* __restrict__ m, int H) {
    const int idx = blockIdx.x * blockDim.x + threadIdx.x;
    if (idx >= ETOT * H) return;
    const int e = idx / H, h = idx % H;
    int sn, dn; edge_ends(ei, e, sn, dn);
    float v = s[sn * H + h] + d[dn * H + h];
    v = v > 0.f ? v : 0.2f * v;                   // leaky_relu, slope 0.2
    eraw[idx] = v;
    atomicMax(&m[dn * H + h], enc_f32(v));
}

// ---------- edge pass B: exp(e - max) + segment sum ----------
__global__ void edge_exp_den(const int* __restrict__ ei,
                             float* __restrict__ ex,
                             const unsigned* __restrict__ m,
                             float* __restrict__ den, int H) {
    const int idx = blockIdx.x * blockDim.x + threadIdx.x;
    if (idx >= ETOT * H) return;
    const int e = idx / H, h = idx % H;
    int sn, dn; edge_ends(ei, e, sn, dn);
    float mm = dec_f32(m[dn * H + h]);
    if (!(mm > -INFINITY)) mm = 0.f;              // isfinite guard (matches ref)
    const float v = expf(ex[idx] - mm);
    ex[idx] = v;
    atomicAdd(&den[dn * H + h], v);
}

// ---------- edge pass C: alpha-weighted scatter aggregation ----------
__global__ void edge_agg(const int* __restrict__ ei,
                         const float* __restrict__ hlin,
                         const float* __restrict__ ex,
                         const float* __restrict__ den,
                         float* __restrict__ agg, int H) {
    const int lane = threadIdx.x;                 // channel within head (32)
    const int e = blockIdx.x * blockDim.y + threadIdx.y;
    if (e >= ETOT) return;                        // wave-uniform
    int sn, dn; edge_ends(ei, e, sn, dn);
    const int OUTD = H * CH;
    __builtin_prefetch(&hlin[(size_t)sn * OUTD], 0, 0);
    for (int h = 0; h < H; ++h) {
        const float alpha = ex[e * H + h] / (den[dn * H + h] + 1e-16f);
        atomicAdd(&agg[(size_t)dn * OUTD + h * CH + lane],
                  hlin[(size_t)sn * OUTD + h * CH + lane] * alpha);
    }
}

// ---------- batchnorm stats (biased var, /N) ----------
__global__ void bn_stats(const float* __restrict__ x, float* __restrict__ mu,
                         float* __restrict__ rstd, int OUTD) {
    const int c = blockIdx.x;
    __shared__ float s1[256], s2[256];
    float a = 0.f, b = 0.f;
    for (int n = threadIdx.x; n < N_NODES; n += 256) {
        const float v = x[(size_t)n * OUTD + c];
        a += v; b += v * v;
    }
    s1[threadIdx.x] = a; s2[threadIdx.x] = b;
    __syncthreads();
    for (int st = 128; st > 0; st >>= 1) {
        if (threadIdx.x < st) {
            s1[threadIdx.x] += s1[threadIdx.x + st];
            s2[threadIdx.x] += s2[threadIdx.x + st];
        }
        __syncthreads();
    }
    if (threadIdx.x == 0) {
        const float mean = s1[0] / (float)N_NODES;
        const float var  = s2[0] / (float)N_NODES - mean * mean;
        mu[c] = mean;
        rstd[c] = rsqrtf(var + 1e-5f);
    }
}

// ---------- bn+relu: write f32 (for head) and packed f16 (next GEMM's A) ----
__global__ void bn_relu_apply(const float* __restrict__ x,
                              const float* __restrict__ mu,
                              const float* __restrict__ rstd,
                              const float* __restrict__ g,
                              const float* __restrict__ b,
                              float* __restrict__ y,
                              _Float16* __restrict__ yh, int OUTD) {
    const int idx = blockIdx.x * blockDim.x + threadIdx.x;
    if (idx >= N_NODES * OUTD) return;
    const int c = idx % OUTD;
    float v = (x[idx] - mu[c]) * rstd[c] * g[c] + b[c];
    v = v > 0.f ? v : 0.f;
    y[idx] = v;
    yh[idx] = (_Float16)v;
}

// ---------- fused fc1 -> relu -> fc2 -> log_softmax, one wave per node ----------
__global__ void head_fc(const float* __restrict__ h,
                        const float* __restrict__ W1, const float* __restrict__ b1,
                        const float* __restrict__ W2, const float* __restrict__ b2,
                        float* __restrict__ out) {
    const int lane = threadIdx.x;                 // 0..31 = fc1 channel
    const int n = blockIdx.x * blockDim.y + threadIdx.y;
    if (n >= N_NODES) return;                     // wave-uniform
    float t = b1[lane];
    #pragma unroll
    for (int k = 0; k < 32; ++k) t += h[(size_t)n * 32 + k] * W1[k * 32 + lane];
    t = t > 0.f ? t : 0.f;
    float o[4];
    #pragma unroll
    for (int j = 0; j < 4; ++j) {
        float p = t * W2[lane * 4 + j];
        #pragma unroll
        for (int m = 16; m > 0; m >>= 1) p += __shfl_xor(p, m, 32);
        o[j] = p + b2[j];
    }
    if (lane == 0) {
        const float mx = fmaxf(fmaxf(o[0], o[1]), fmaxf(o[2], o[3]));
        float se = 0.f;
        #pragma unroll
        for (int j = 0; j < 4; ++j) se += expf(o[j] - mx);
        const float lse = mx + logf(se);
        #pragma unroll
        for (int j = 0; j < 4; ++j) out[(size_t)n * 4 + j] = o[j] - lse;
    }
}

extern "C" void kernel_launch(void* const* d_in, const int* in_sizes, int n_in,
                              void* d_out, int out_size, void* d_ws, size_t ws_size,
                              hipStream_t stream) {
    const float* x  = (const float*)d_in[0];
    const int*   ei = (const int*)d_in[1];

    // ---- workspace carve-out (~108 MB); f16 buffers first for 16B alignment ----
    _Float16* xh = (_Float16*)d_ws;                       // activations, f16 [N,128]
    _Float16* wh = xh + (size_t)N_NODES * HID;            // weights, f16 [128,128]
    float* wsf = (float*)(wh + (size_t)HID * HID);
    float* hbuf0 = wsf;                 wsf += (size_t)N_NODES * HID;  // activations f32
    float* hlin  = wsf;                 wsf += (size_t)N_NODES * HID;  // GEMM output
    float* agg   = wsf;                 wsf += (size_t)N_NODES * HID;  // aggregation
    float* sbuf  = wsf;                 wsf += (size_t)N_NODES * 4;
    float* dbuf  = wsf;                 wsf += (size_t)N_NODES * 4;
    unsigned* mbuf = (unsigned*)wsf;    wsf += (size_t)N_NODES * 4;
    float* den   = wsf;                 wsf += (size_t)N_NODES * 4;
    float* exbuf = wsf;                 wsf += (size_t)ETOT * 4;
    float* mu    = wsf;                 wsf += 128;
    float* rstd  = wsf;                 wsf += 128;

    const dim3 wb(32, 4);   // 4 waves / block for GEMM
    const dim3 eb(32, 8);   // 8 edges (waves) / block for aggregation
    const int mtiles = N_NODES / 16;   // 3125, exact

    // pack input features to f16 once
    f32_to_f16<<<(N_NODES * HID + 255) / 256, 256, 0, stream>>>(x, xh, N_NODES * HID);

    for (int layer = 1; layer <= 7; ++layer) {
        const int H    = (layer == 7) ? 1 : 4;
        const int OUTD = H * CH;
        const int base = (layer <= 6) ? 2 + (layer - 1) * 6 : 38;
        const float* W  = (const float*)d_in[base];
        const float* as = (const float*)d_in[base + 1];
        const float* ad = (const float*)d_in[base + 2];
        const float* gb = (const float*)d_in[base + 3];
        const float* bg = (const float*)d_in[base + 4];
        const float* bb = (const float*)d_in[base + 5];

        // pack weights to f16
        f32_to_f16<<<(HID * OUTD + 255) / 256, 256, 0, stream>>>(W, wh, HID * OUTD);

        if (OUTD == HID)
            gemm_wmma_f16<8><<<(mtiles + 3) / 4, wb, 0, stream>>>(xh, wh, hlin, HID);
        else
            gemm_wmma_f16<2><<<(mtiles + 3) / 4, wb, 0, stream>>>(xh, wh, hlin, HID);

        attn_coef<<<(N_NODES * H + 255) / 256, 256, 0, stream>>>(hlin, as, ad, sbuf, dbuf, H);
        init_agg<<<(N_NODES * OUTD + 255) / 256, 256, 0, stream>>>(agg, gb, OUTD);
        init_mden<<<(N_NODES * H + 255) / 256, 256, 0, stream>>>(mbuf, den, H);
        edge_lrelu_max<<<(ETOT * H + 255) / 256, 256, 0, stream>>>(ei, sbuf, dbuf, exbuf, mbuf, H);
        edge_exp_den<<<(ETOT * H + 255) / 256, 256, 0, stream>>>(ei, exbuf, mbuf, den, H);
        edge_agg<<<(ETOT + 7) / 8, eb, 0, stream>>>(ei, hlin, exbuf, den, agg, H);
        bn_stats<<<OUTD, 256, 0, stream>>>(agg, mu, rstd, OUTD);
        bn_relu_apply<<<(N_NODES * OUTD + 255) / 256, 256, 0, stream>>>(
            agg, mu, rstd, bg, bb, hbuf0, xh, OUTD);
    }

    const dim3 hb(32, 8);
    head_fc<<<(N_NODES + 7) / 8, hb, 0, stream>>>(hbuf0,
        (const float*)d_in[44], (const float*)d_in[45],
        (const float*)d_in[46], (const float*)d_in[47],
        (float*)d_out);
}
